// DiffusionActionHead_19980187861206
// MI455X (gfx1250) — matrix operand order repaired and checked by reading, not compiled, not run on hardware
//
#include <hip/hip_runtime.h>

// Problem constants (from the reference)
#define NB    32      // B
#define NT    32      // T
#define SDIM  64
#define ADIM  32
#define HID   1024
#define EMB   1536    // N of every category GEMM (compile-time!)
#define EMB2  3072

#define WROW  40      // padded LDS row stride (bf16 elems): 80B, 16B-aligned,
                      // breaks the 64B-row == 64-bank store conflict (32-way -> 8-way)

typedef __attribute__((ext_vector_type(16))) __bf16       v16bf;
typedef __attribute__((ext_vector_type(8)))  float        v8f;
typedef __attribute__((ext_vector_type(4)))  unsigned int v4u;

union Frag16 { v16bf v; v4u u[2]; };
union PackA  { __bf16 h[4]; uint2 u; };

// sinusoidal positional encoding element j of a 1536-wide embedding
__device__ __forceinline__ float tau_val(float t, int j) {
    const int   i = (j < 768) ? j : j - 768;
    const float d = __expf(-9.210340371976184f * (float)i * (1.0f / 768.0f)); // ln(1e4)
    const float a = t * d;
    return (j < 768) ? __sinf(a) : __cosf(a);
}

// ---------------------------------------------------------------------------
// Category-specific GEMM:  C[b] (32 x EMB) = act( A[b] (32 x K) @ W[cat[b]] + bias )
// TAU: columns K_A..K-1 of A are the sinusoid tau embedding, computed on the
// fly (identical for all 32 rows; slab-uniform branch since K_A % 32 == 0).
// Block: 256 threads (8 waves), one 128-column N tile of one batch element.
// Each wave: 32x16 tile = two v_wmma_f32_16x16x32_bf16 per K-step.
// Software pipeline: global->regs (k+32) | wmma (k) | regs->LDS (k+32) | bar.
// ---------------------------------------------------------------------------
template <int ACT, bool TAU>
__global__ __launch_bounds__(256)
void gemm_cat(const float* __restrict__ A, const float* __restrict__ W,
              const float* __restrict__ bias, const int* __restrict__ cat,
              const int* __restrict__ tstep, float* __restrict__ C,
              int K, int K_A, int c_batch_stride, int c_row_off) {
    const int tid  = threadIdx.x;
    const int lane = tid & 31;
    const int wv   = tid >> 5;
    const int b    = blockIdx.y;
    const int n0   = blockIdx.x * 128;
    const int c    = cat[b];

    __shared__ __bf16 Ash[2][32 * WROW];    // [row][k]        2 x 2.5 KB
    __shared__ __bf16 Wsh[2][128 * WROW];   // [n][k] transp.  2 x 10 KB

    const float* Wc = W + (size_t)c * K * EMB + n0;   // 64-bit once, outside loop
    const float* Ab = A + b * (32 * K_A);             // <= 1.5M elems: int math
    const float  tb = TAU ? (float)tstep[b] : 0.0f;

    // per-thread staging coordinates (all int)
    const int a_row = tid >> 3;               // 0..31
    const int a_k   = (tid & 7) * 4;          // 0,4,..,28
    const int w_k   = tid >> 5;               // 0..7  (k rows, +8 per chunk)
    const int w_n   = (tid & 31) * 4;         // 0..124 (4 consecutive n)
    const int w_off = w_k * EMB + w_n;

    float4 rA, rW0, rW1, rW2, rW3;

    // ---- pipeline stages (macros so locals stay in registers) -----------
#define LG(KT_NEXT)                                                          \
    {                                                                        \
        const int ktn_ = (KT_NEXT);                                          \
        if (TAU && ktn_ >= K_A) {                                            \
            const int j_ = ktn_ - K_A + a_k;                                 \
            rA.x = tau_val(tb, j_);     rA.y = tau_val(tb, j_ + 1);          \
            rA.z = tau_val(tb, j_ + 2); rA.w = tau_val(tb, j_ + 3);          \
        } else {                                                             \
            rA = *(const float4*)(Ab + a_row * K_A + ktn_ + a_k);            \
        }                                                                    \
        const float* wp_ = Wc + ktn_ * EMB + w_off;                          \
        rW0 = *(const float4*)(wp_);                                         \
        rW1 = *(const float4*)(wp_ + 8  * EMB);                              \
        rW2 = *(const float4*)(wp_ + 16 * EMB);                              \
        rW3 = *(const float4*)(wp_ + 24 * EMB);                              \
    }

#define STW(P_, RV, I)                                                       \
    Wsh[P_][(w_n + 0) * WROW + (w_k + 8 * (I))] = (__bf16)RV.x;              \
    Wsh[P_][(w_n + 1) * WROW + (w_k + 8 * (I))] = (__bf16)RV.y;              \
    Wsh[P_][(w_n + 2) * WROW + (w_k + 8 * (I))] = (__bf16)RV.z;              \
    Wsh[P_][(w_n + 3) * WROW + (w_k + 8 * (I))] = (__bf16)RV.w;

#define SL(P_)                                                               \
    {                                                                        \
        PackA pa_;                                                           \
        pa_.h[0] = (__bf16)rA.x; pa_.h[1] = (__bf16)rA.y;                    \
        pa_.h[2] = (__bf16)rA.z; pa_.h[3] = (__bf16)rA.w;                    \
        *(uint2*)&Ash[P_][a_row * WROW + a_k] = pa_.u;                       \
        STW(P_, rW0, 0) STW(P_, rW1, 1) STW(P_, rW2, 2) STW(P_, rW3, 3)      \
    }

    v8f acc0 = {};
    v8f acc1 = {};

    LG(0);
    SL(0);
    __syncthreads();

    int p = 0;
    for (int kt = 0; kt < K; kt += 32, p ^= 1) {
        const bool more = (kt + 32 < K);
        if (more) LG(kt + 32);   // long-latency global loads issue first

        // fragments from buffer p (documented gfx1250 16-bit layouts)
        Frag16 fb, fa0, fa1;
        {
            // B (32x16): lanes 0-15 K {0..7,16..23} of col n, lanes 16-31 K {8..15,24..31}
            const __bf16* wp = &Wsh[p][((wv << 4) + (lane & 15)) * WROW +
                                       ((lane >> 4) << 3)];
            fb.u[0] = *(const v4u*)(wp);
            fb.u[1] = *(const v4u*)(wp + 16);
            // A (16x32): lane<16 -> row=lane, K {0..7,16..23}; lane>=16 -> K {8..15,24..31}
            const int r  = lane & 15;
            const int kh = (lane >> 4) << 3;
            const __bf16* ap0 = &Ash[p][r * WROW + kh];
            fa0.u[0] = *(const v4u*)(ap0);
            fa0.u[1] = *(const v4u*)(ap0 + 16);
            const __bf16* ap1 = &Ash[p][(16 + r) * WROW + kh];
            fa1.u[0] = *(const v4u*)(ap1);
            fa1.u[1] = *(const v4u*)(ap1 + 16);
        }
        acc0 = __builtin_amdgcn_wmma_f32_16x16x32_bf16(false, fa0.v, false, fb.v,
                                                       (short)0, acc0, false, false);
        acc1 = __builtin_amdgcn_wmma_f32_16x16x32_bf16(false, fa1.v, false, fb.v,
                                                       (short)0, acc1, false, false);

        if (more) {
            SL(p ^ 1);
            if (tid < 32 && kt + 64 < K)           // emits global_prefetch_b8
                __builtin_prefetch(Wc + (kt + 64 + tid) * EMB, 0, 1);
        }
        __syncthreads();
    }
#undef LG
#undef SL
#undef STW

    // ---- epilogue: bias + activation + store ----------------------------
    // C/D layout: VGPR v, lane<16 -> M=v, N=lane ; lane>=16 -> M=v+8, N=lane-16
    const int   n     = n0 + (wv << 4) + (lane & 15);
    const float bv    = bias[c * EMB + n];
    const int   rbase = (lane >> 4) << 3;
    float* Cb = C + (size_t)b * c_batch_stride + c_row_off * EMB;
    #pragma unroll
    for (int v = 0; v < 8; ++v) {
        float x0 = acc0[v] + bv;
        float x1 = acc1[v] + bv;
        if (ACT == 1) {  // swish: x * sigmoid(x)
            x0 = x0 / (1.0f + __expf(-x0));
            x1 = x1 / (1.0f + __expf(-x1));
        }
        Cb[(rbase + v) * EMB + n]        = x0;
        Cb[(16 + rbase + v) * EMB + n]   = x1;
    }
}

// ---------------------------------------------------------------------------
// State encoder layer 1: h[b] = relu(state[b] (1x64) @ se_W1[c] + se_b1[c])
// M=1 GEMV, float4-vectorized along the output dim.
// ---------------------------------------------------------------------------
__global__ __launch_bounds__(256)
void se1_kernel(const float* __restrict__ state, const float* __restrict__ W1,
                const float* __restrict__ b1, const int* __restrict__ cat,
                float* __restrict__ h) {
    const int b = blockIdx.x;
    const int tid = threadIdx.x;
    const int j = tid * 4;                     // 256 threads x 4 = 1024 = HID
    const int c = cat[b];
    __shared__ float s[SDIM];
    if (tid < SDIM) s[tid] = state[b * SDIM + tid];
    __syncthreads();
    const float* W = W1 + (size_t)c * (SDIM * HID) + j;
    float4 sum = *(const float4*)(b1 + c * HID + j);
    #pragma unroll 8
    for (int k = 0; k < SDIM; ++k) {
        const float4 w = *(const float4*)(W + k * HID);
        const float  hv = s[k];
        sum.x = fmaf(hv, w.x, sum.x);
        sum.y = fmaf(hv, w.y, sum.y);
        sum.z = fmaf(hv, w.z, sum.z);
        sum.w = fmaf(hv, w.w, sum.w);
    }
    sum.x = fmaxf(sum.x, 0.0f); sum.y = fmaxf(sum.y, 0.0f);
    sum.z = fmaxf(sum.z, 0.0f); sum.w = fmaxf(sum.w, 0.0f);
    *(float4*)(h + b * HID + j) = sum;
}

// ---------------------------------------------------------------------------
// State encoder layer 2: out[b,0,:] = h[b] (1x1024) @ se_W2[c] + se_b2[c]
// 201 MB weight stream -> float4 loads, 4 output columns per thread.
// ---------------------------------------------------------------------------
__global__ __launch_bounds__(128)
void se2_kernel(const float* __restrict__ h, const float* __restrict__ W2,
                const float* __restrict__ b2, const int* __restrict__ cat,
                float* __restrict__ out) {
    const int b = blockIdx.y;
    const int tid = threadIdx.x;
    const int j = (blockIdx.x * 128 + tid) * 4;   // 3 blocks x 128 x 4 = 1536
    const int c = cat[b];
    __shared__ float hs[HID];
    #pragma unroll
    for (int i = 0; i < 8; ++i) hs[tid + i * 128] = h[b * HID + tid + i * 128];
    __syncthreads();
    const float* W = W2 + (size_t)c * (HID * EMB) + j;
    float4 sum = *(const float4*)(b2 + c * EMB + j);
    #pragma unroll 4
    for (int k = 0; k < HID; ++k) {
        const float4 w = *(const float4*)(W + k * EMB);
        const float  hv = hs[k];
        sum.x = fmaf(hv, w.x, sum.x);
        sum.y = fmaf(hv, w.y, sum.y);
        sum.z = fmaf(hv, w.z, sum.z);
        sum.w = fmaf(hv, w.w, sum.w);
    }
    *(float4*)(out + (size_t)b * ((NT + 1) * EMB) + j) = sum;  // row 0 = state token
}

// ---------------------------------------------------------------------------
extern "C" void kernel_launch(void* const* d_in, const int* in_sizes, int n_in,
                              void* d_out, int out_size, void* d_ws, size_t ws_size,
                              hipStream_t stream) {
    const float* state   = (const float*)d_in[0];
    const float* actions = (const float*)d_in[1];
    const int*   tsteps  = (const int*)d_in[2];
    const int*   cat     = (const int*)d_in[3];
    const float* se_W1   = (const float*)d_in[4];
    const float* se_b1   = (const float*)d_in[5];
    const float* se_W2   = (const float*)d_in[6];
    const float* se_b2   = (const float*)d_in[7];
    const float* ae_W1   = (const float*)d_in[8];
    const float* ae_b1   = (const float*)d_in[9];
    const float* ae_W2   = (const float*)d_in[10];
    const float* ae_b2   = (const float*)d_in[11];
    const float* ae_W3   = (const float*)d_in[12];
    const float* ae_b3   = (const float*)d_in[13];
    float* out = (float*)d_out;

    // workspace: X (B*32*EMB) | X2 (B*32*EMB) | h (B*HID)  ~= 12.7 MB
    float* X  = (float*)d_ws;
    float* X2 = X + (size_t)NB * NT * EMB;
    float* h  = X2 + (size_t)NB * NT * EMB;

    // state encoder (M=1 path)
    se1_kernel<<<dim3(NB), 256, 0, stream>>>(state, se_W1, se_b1, cat, h);
    se2_kernel<<<dim3(3, NB), 128, 0, stream>>>(h, se_W2, se_b2, cat, out);

    // ae layer 1: X = actions @ W1 + b1            (K=32)
    gemm_cat<0, false><<<dim3(EMB / 128, NB), 256, 0, stream>>>(
        actions, ae_W1, ae_b1, cat, nullptr, X, ADIM, ADIM, NT * EMB, 0);

    // ae layer 2: X2 = swish([X | tau] @ W2 + b2)  (K=3072, tau on the fly)
    gemm_cat<1, true><<<dim3(EMB / 128, NB), 256, 0, stream>>>(
        X, ae_W2, ae_b2, cat, tsteps, X2, EMB2, EMB, NT * EMB, 0);

    // ae layer 3: out[:,1:,:] = X2 @ W3 + b3       (K=1536)
    gemm_cat<0, false><<<dim3(EMB / 128, NB), 256, 0, stream>>>(
        X2, ae_W3, ae_b3, cat, nullptr, out, EMB, EMB, (NT + 1) * EMB, 1);
}
